// SelfAttention_2628519985488
// MI455X (gfx1250) — matrix operand order repaired and checked by reading, not compiled, not run on hardware
//
#include <hip/hip_runtime.h>
#include <hip/hip_bf16.h>

#define BDIM 4
#define CDIM 256
#define C8   32
#define LDIM 4096
#define VPAD 72          // padded LDS row stride (bf16 elems): 36 words, conflict-free

typedef __attribute__((ext_vector_type(16))) __bf16 v16bf;
typedef __attribute__((ext_vector_type(8)))  __bf16 v8bf;
typedef __attribute__((ext_vector_type(8)))  float  v8f;

static __device__ __forceinline__ v16bf ld16bf(const __bf16* p0, const __bf16* p1) {
  union { v16bf v; v8bf h[2]; } u;
  u.h[0] = *(const v8bf*)p0;
  u.h[1] = *(const v8bf*)p1;
  return u.v;
}

// ---------------------------------------------------------------------------
// Phase 1: pointwise conv projection  y = W x + b, f32 -> bf16.
// transposed=1 : y[b][l][o]   (q/k, O=32 : WMMA K-dim layout)
// transposed=0 : y[b][o][l]   (v,   O=256: row-major for A-operand rows)
// ---------------------------------------------------------------------------
__global__ __launch_bounds__(128) void proj_kernel(
    const float* __restrict__ x, const float* __restrict__ W,
    const float* __restrict__ bias, __bf16* __restrict__ y,
    int Odim, int transposed)
{
  __shared__ float xs[64 * 128];
  const int tid = threadIdx.x;
  const int l0  = blockIdx.x * 128;
  const int b   = blockIdx.y;
  const int o0  = blockIdx.z * 32;

  float acc[32];
  #pragma unroll
  for (int o = 0; o < 32; ++o) acc[o] = 0.f;

  for (int cc = 0; cc < CDIM; cc += 64) {
    __syncthreads();
    #pragma unroll 8
    for (int r = 0; r < 64; ++r)
      xs[r * 128 + tid] = x[(size_t)(b * CDIM + cc + r) * LDIM + l0 + tid];
    __syncthreads();
    for (int r = 0; r < 64; ++r) {
      float xv = xs[r * 128 + tid];
      #pragma unroll
      for (int o = 0; o < 32; ++o)
        acc[o] = fmaf(W[(o0 + o) * CDIM + cc + r], xv, acc[o]);   // uniform s_load
    }
  }

  if (transposed) {
    #pragma unroll
    for (int o = 0; o < 32; ++o)
      y[((size_t)b * LDIM + l0 + tid) * (size_t)Odim + o0 + o] =
          (__bf16)(acc[o] + bias[o0 + o]);
  } else {
    #pragma unroll
    for (int o = 0; o < 32; ++o)
      y[(size_t)(b * Odim + o0 + o) * LDIM + l0 + tid] =
          (__bf16)(acc[o] + bias[o0 + o]);
  }
}

// ---------------------------------------------------------------------------
// Phase 2: flash attention, transposed orientation, V shared through LDS.
//   S^T = K^T Q  (M=j, N=i)  -> per-lane softmax state (i = lane&15)
//   O^T = V  P^T (M=c, N=i)  -> per-lane rescale, coalesced epilogue over i
// Block = 256 thr (8 waves) = 4 i-subtiles (64 i rows) x 2 channel halves.
// V tile (256c x 64j) double-buffered in LDS: 4x less HBM traffic on V.
// ---------------------------------------------------------------------------
__global__ __launch_bounds__(256) void attn_kernel(
    const __bf16* __restrict__ qT, const __bf16* __restrict__ kT,
    const __bf16* __restrict__ vmat, const float* __restrict__ x,
    const float* __restrict__ gamma, float* __restrict__ out)
{
  __shared__ __bf16 vs[2][CDIM * VPAD];   // 2 x 36 KB V tiles
  __shared__ __bf16 pbuf[8][16 * VPAD];   // per-wave P^T staging, 18 KB

  const int tid  = threadIdx.x;
  const int lane = tid & 31;
  const int wave = tid >> 5;
  const int it   = wave & 3;              // i-subtile within block
  const int ch   = wave >> 2;             // channel half (0/1)
  const int b    = blockIdx.y;
  const int i0   = blockIdx.x * 64 + it * 16;
  const int c0   = ch * 128;

  const int li  = lane & 15;
  const int hi  = lane >> 4;              // 0: lanes 0-15, 1: lanes 16-31
  const int akb = hi ? 8 : 0;             // A-operand per-lane K chunk base
  const int bkb = hi ? 8 : 0;             // B-operand per-lane K base

  // B-operand Q: (K=c 32) x (N=i 16), loaded once. Column i = i0+li.
  const __bf16* qrow = qT + ((size_t)b * LDIM + i0 + li) * C8 + bkb;
  const v16bf Qb = ld16bf(qrow, qrow + 16);

  v8f O[8];
  #pragma unroll
  for (int ct = 0; ct < 8; ++ct)
    #pragma unroll
    for (int r = 0; r < 8; ++r) O[ct][r] = 0.f;

  float m = -INFINITY, lsum = 0.f;

  v8f zs;
  #pragma unroll
  for (int r = 0; r < 8; ++r) zs[r] = 0.f;

  // V staging: thread tid owns channel row tid (64 j = 128 B = 8 x b128)
  const __bf16* vrow = vmat + ((size_t)b * CDIM + tid) * LDIM;
  v8bf st[8];
  #pragma unroll
  for (int k = 0; k < 8; ++k) st[k] = *(const v8bf*)(vrow + k * 8);
  {
    __bf16* q = &vs[0][tid * VPAD];
    #pragma unroll
    for (int k = 0; k < 8; ++k) *(v8bf*)(q + k * 8) = st[k];
  }

  const int NS = LDIM / 64;
  for (int js = 0; js < NS; ++js) {
    const int j0 = js * 64;

    // issue global loads for next V slab (overlap with compute below)
    if (js + 1 < NS) {
      const __bf16* p = vrow + j0 + 64;
      #pragma unroll
      for (int k = 0; k < 8; ++k) st[k] = *(const v8bf*)(p + k * 8);
      __builtin_prefetch(kT + ((size_t)b * LDIM + j0 + 64 + li) * C8, 0, 3);
    }

    __syncthreads();   // vs[js&1] ready; vs[(js+1)&1] free to overwrite

    // ---- S^T tiles: A = kT rows (M=j, K=c), B = Q ----
    v8f STt[4];
    #pragma unroll
    for (int t = 0; t < 4; ++t) {
      const __bf16* krow = kT + ((size_t)b * LDIM + j0 + t * 16 + li) * C8;
      v16bf Ak = ld16bf(krow + akb, krow + akb + 16);
      STt[t] = __builtin_amdgcn_wmma_f32_16x16x32_bf16(false, Ak, false, Qb,
                                                       (short)0, zs, false, false);
    }

    // ---- per-lane online softmax (row i = i0+li) ----
    float tm = STt[0][0];
    #pragma unroll
    for (int t = 0; t < 4; ++t)
      #pragma unroll
      for (int r = 0; r < 8; ++r) tm = fmaxf(tm, STt[t][r]);
    tm = fmaxf(tm, __shfl_xor(tm, 16, 32));       // merge j-halves (lane^16)

    const float mn   = fmaxf(m, tm);
    const float corr = __expf(m - mn);
    m = mn;

    __bf16* pw = &pbuf[wave][0] + li * VPAD;      // row i, 64 j-slots
    float rs = 0.f;
    #pragma unroll
    for (int t = 0; t < 4; ++t)
      #pragma unroll
      for (int r = 0; r < 8; ++r) {
        float p = __expf(STt[t][r] - mn);
        rs += p;
        pw[t * 16 + r + 8 * hi] = (__bf16)p;      // P^T staged as [i][j]
      }
    rs += __shfl_xor(rs, 16, 32);
    lsum = lsum * corr + rs;

    #pragma unroll
    for (int ct = 0; ct < 8; ++ct)
      #pragma unroll
      for (int r = 0; r < 8; ++r) O[ct][r] *= corr;

    asm volatile("s_wait_dscnt 0" ::: "memory");

    // ---- B-operands P^T: two K=j slabs of 32 ----
    const __bf16* pr = &pbuf[wave][0] + li * VPAD;
    const v16bf Pb0 = ld16bf(pr + bkb, pr + bkb + 16);
    const v16bf Pb1 = ld16bf(pr + 32 + bkb, pr + 32 + bkb + 16);

    // ---- O^T += V (M=c, K=j, from LDS) * P^T (K=j, N=i) ----
    const __bf16* vb = &vs[js & 1][0];
    #pragma unroll
    for (int ct = 0; ct < 8; ++ct) {
      const __bf16* vr = vb + (c0 + ct * 16 + li) * VPAD;
      v16bf Va0 = ld16bf(vr + akb, vr + akb + 16);
      v16bf Va1 = ld16bf(vr + 32 + akb, vr + 32 + akb + 16);
      O[ct] = __builtin_amdgcn_wmma_f32_16x16x32_bf16(false, Va0, false, Pb0,
                                                      (short)0, O[ct], false, false);
      O[ct] = __builtin_amdgcn_wmma_f32_16x16x32_bf16(false, Va1, false, Pb1,
                                                      (short)0, O[ct], false, false);
    }

    // stage next V slab into the other buffer (reads of it ended pre-barrier)
    if (js + 1 < NS) {
      __bf16* q = &vs[(js + 1) & 1][tid * VPAD];
      #pragma unroll
      for (int k = 0; k < 8; ++k) *(v8bf*)(q + k * 8) = st[k];
    }
  }

  // ---- epilogue: normalize, gamma * O + x (coalesced over i = lanes) ----
  const float g   = gamma[0];
  const float inv = 1.f / lsum;
  const int   i   = i0 + li;
  #pragma unroll
  for (int ct = 0; ct < 8; ++ct)
    #pragma unroll
    for (int r = 0; r < 8; ++r) {
      int c = c0 + ct * 16 + r + 8 * hi;
      size_t idx = (size_t)(b * CDIM + c) * LDIM + i;
      out[idx] = g * (O[ct][r] * inv) + x[idx];
    }
}

// ---------------------------------------------------------------------------
extern "C" void kernel_launch(void* const* d_in, const int* in_sizes, int n_in,
                              void* d_out, int out_size, void* d_ws, size_t ws_size,
                              hipStream_t stream) {
  const float* x     = (const float*)d_in[0];
  const float* Wq    = (const float*)d_in[1];
  const float* bq    = (const float*)d_in[2];
  const float* Wk    = (const float*)d_in[3];
  const float* bk    = (const float*)d_in[4];
  const float* Wv    = (const float*)d_in[5];
  const float* bv    = (const float*)d_in[6];
  const float* gamma = (const float*)d_in[7];
  float* out = (float*)d_out;

  char* ws = (char*)d_ws;
  const size_t qk_bytes = (size_t)BDIM * LDIM * C8 * 2;     // 1 MB each
  __bf16* qT = (__bf16*)ws;
  __bf16* kT = (__bf16*)(ws + qk_bytes);
  __bf16* vW = (__bf16*)(ws + 2 * qk_bytes);                // 8 MB

  proj_kernel<<<dim3(LDIM / 128, BDIM, 1), dim3(128), 0, stream>>>(x, Wq, bq, qT, C8, 1);
  proj_kernel<<<dim3(LDIM / 128, BDIM, 1), dim3(128), 0, stream>>>(x, Wk, bk, kT, C8, 1);
  proj_kernel<<<dim3(LDIM / 128, BDIM, CDIM / 32), dim3(128), 0, stream>>>(x, Wv, bv, vW, CDIM, 0);
  attn_kernel<<<dim3(LDIM / 64, BDIM), dim3(256), 0, stream>>>(qT, kT, vW, x, gamma, out);
}
